// NATGAT_76836964926279
// MI455X (gfx1250) — matrix operand order repaired and checked by reading, not compiled, not run on hardware
//
#include <hip/hip_runtime.h>
#include <hip/hip_bf16.h>

#define NNODES 16384
#define NEDGES 65536
#define KSEQ   8
#define DIM    128
#define HEADS  8
#define HD     16      // DIM / HEADS
#define NHT    4       // HEADS / 2
#define HDT    32      // DIM / NHT
#define NWPACK 9       // packed 128x128 weights
#define WPELEM 16384   // halves per packed weight (4 kb * 8 n0 * 32 lanes * 16)

typedef _Float16 v16h __attribute__((ext_vector_type(16)));
typedef float    v8f  __attribute__((ext_vector_type(8)));
typedef _Float16 h4v  __attribute__((ext_vector_type(4)));

__device__ __forceinline__ v8f wmma16(v16h a, v16h b, v8f c) {
  return __builtin_amdgcn_wmma_f32_16x16x32_f16(false, a, false, b, (short)0, c,
                                                false, false);
}

__device__ __forceinline__ h4v f4toh4(float4 v) {
  h4v h;
  h[0] = (_Float16)v.x; h[1] = (_Float16)v.y;
  h[2] = (_Float16)v.z; h[3] = (_Float16)v.w;
  return h;
}

// A fragment (16x32 f16) from a 16x128 f16 LDS tile.
// lanes0-15 M=lane, v0..3 K=0..7, v4..7 K=16..23; lanes16-31 K+=8.
// Per half-fragment the 8 halves are contiguous -> compiler emits ds_load_b128.
__device__ __forceinline__ v16h load_afrag(const _Float16* at, int kb, int lane) {
  const int m = lane & 15, hi = lane >> 4;
  v16h a;
#pragma unroll
  for (int i = 0; i < 16; ++i) {
    int k = kb + ((i >> 3) << 4) + (hi << 3) + (((i >> 1) & 3) << 1) + (i & 1);
    a[i] = at[m * DIM + k];
  }
  return a;
}

// One wave: 16x16 output slice, full K=128 reduction (4 WMMA).
// B fragments come from pre-packed f16 weights: one aligned 32B load per frag.
__device__ __forceinline__ v8f tile_gemm_p(const _Float16* at, const _Float16* pw,
                                           int wave, int lane) {
  v8f acc = {};
#pragma unroll
  for (int kb = 0; kb < DIM; kb += 32) {
    v16h a = load_afrag(at, kb, lane);
    v16h b = *(const v16h*)(pw + (((kb >> 5) << 3) + wave) * 512 + lane * 16);
    acc = wmma16(a, b, acc);
  }
  return acc;
}

__device__ __forceinline__ void store_tile(float* out, int row0, v8f acc,
                                           const float* bias, int wave, int lane) {
  const int n = (wave << 4) + (lane & 15), hi = lane >> 4;
  const float bv = bias[n];
#pragma unroll
  for (int v = 0; v < 8; ++v)
    out[(size_t)(row0 + v + (hi << 3)) * DIM + n] = acc[v] + bv;
}

// sortable-uint encoding of float for atomic max
__device__ __forceinline__ unsigned f2s(float f) {
  unsigned u = __float_as_uint(f);
  return (u & 0x80000000u) ? ~u : (u | 0x80000000u);
}
__device__ __forceinline__ float s2f(unsigned s) {
  unsigned u = (s & 0x80000000u) ? (s & 0x7FFFFFFFu) : ~s;
  return __uint_as_float(u);
}
__device__ __forceinline__ float geluf(float s) {
  return 0.5f * s * (1.f + erff(s * 0.70710678f));
}
__device__ __forceinline__ float sigm(float s) { return 1.f / (1.f + expf(-s)); }

// ---------------- Kernel 0: pack 9 weights into WMMA B-fragment layout -------
__global__ __launch_bounds__(256) void k_pack(
    const float* Wq, const float* Wk, const float* Wv, const float* Wo,
    const float* Wek, const float* Wev, const float* Wtk,
    const float* tsWq, const float* tsWk, _Float16* wp) {
  int id = blockIdx.x * 256 + threadIdx.x;
  if (id >= NWPACK * WPELEM) return;
  int w = id >> 14, rem = id & (WPELEM - 1);
  int i = rem & 15, lane = (rem >> 4) & 31, nIdx = (rem >> 9) & 7, kbIdx = rem >> 12;
  const float* W;
  switch (w) {
    case 0: W = Wq;  break; case 1: W = Wk;  break; case 2: W = Wv;  break;
    case 3: W = Wo;  break; case 4: W = Wek; break; case 5: W = Wev; break;
    case 6: W = Wtk; break; case 7: W = tsWq; break; default: W = tsWk; break;
  }
  int hi = lane >> 4;
  wp[id] = (_Float16)W[(kbIdx * 32 + ((i >> 3) << 4) + hi * 8 + (i & 7)) * DIM + nIdx * 16 + (lane & 15)];
}

// ---------------- Kernel 1: Q/K/V/g over nodes ----------------
__global__ __launch_bounds__(256) void k_node_qkvg(
    const float* __restrict__ x, const _Float16* pWq, const float* bq,
    const _Float16* pWk, const float* bk, const _Float16* pWv, const float* bv,
    const float* Wg, const float* bg, float* Qo, float* Ko, float* Vo, float* go) {
  __shared__ __align__(16) _Float16 at[16 * DIM];
  const int t = threadIdx.x, lane = t & 31, wave = t >> 5;
  const int row0 = blockIdx.x * 16;
  const float4* xs = (const float4*)(x + (size_t)row0 * DIM);
  for (int j = t; j < 512; j += 256) *(h4v*)(at + 4 * j) = f4toh4(xs[j]);
  __syncthreads();
  store_tile(Qo, row0, tile_gemm_p(at, pWq, wave, lane), bq, wave, lane);
  store_tile(Ko, row0, tile_gemm_p(at, pWk, wave, lane), bk, wave, lane);
  store_tile(Vo, row0, tile_gemm_p(at, pWv, wave, lane), bv, wave, lane);
  if (t < 16) {
    float s = bg[0];
    for (int d = 0; d < DIM; ++d) s += (float)at[t * DIM + d] * Wg[d];
    go[row0 + t] = sigm(s);
  }
}

// ---------------- Kernel 2: edge keys/values ----------------
__global__ __launch_bounds__(256) void k_edge_kv(
    const int* ei, const float* edge_emb, const float* time_emb,
    const _Float16* pWek, const float* bek, const _Float16* pWev, const float* bev,
    const _Float16* pWtk, const float* btk, const float* Kb, const float* Vb,
    float* Kj, float* Vj) {
  __shared__ __align__(16) _Float16 ae[16 * DIM];
  __shared__ __align__(16) _Float16 atm[16 * DIM];
  const int t = threadIdx.x, lane = t & 31, wave = t >> 5;
  const int e0 = blockIdx.x * 16;
  const float4* es = (const float4*)(edge_emb + (size_t)e0 * DIM);
  const float4* ts = (const float4*)(time_emb + (size_t)e0 * DIM);
  for (int j = t; j < 512; j += 256) {
    *(h4v*)(ae + 4 * j)  = f4toh4(es[j]);
    *(h4v*)(atm + 4 * j) = f4toh4(ts[j]);
  }
  __syncthreads();
  v8f ek = tile_gemm_p(ae, pWek, wave, lane);
  v8f ev = tile_gemm_p(ae, pWev, wave, lane);
  v8f tk = tile_gemm_p(atm, pWtk, wave, lane);
  const int n = (wave << 4) + (lane & 15), hi = lane >> 4;
  const float bk2 = bek[n] + btk[n], bv2 = bev[n];
#pragma unroll
  for (int v = 0; v < 8; ++v) {
    int e = e0 + v + (hi << 3);
    int s = ei[e];  // src
    Kj[(size_t)e * DIM + n] = Kb[(size_t)s * DIM + n] + ek[v] + tk[v] + bk2;
    Vj[(size_t)e * DIM + n] = Vb[(size_t)s * DIM + n] + ev[v] + bv2;
  }
}

// ---------------- Kernel 3: trust (heavy WMMA: gathered h_seq GEMMs) ----------
__global__ __launch_bounds__(256) void k_trust(
    const int* ei, const float* x, const float* edge_emb, const float* hseq,
    const _Float16* ptsWq, const float* tsbq, const _Float16* ptsWk, const float* tsbk,
    const float* epW1, const float* epb1, const float* epW2, const float* epb2,
    const float* opW1, const float* opb1, const float* opW2, const float* opb2,
    float* trust) {
  __shared__ __align__(16) _Float16 at[16 * DIM];
  __shared__ float qt[16 * DIM];
  __shared__ float kt[16 * DIM];
  __shared__ float sc[2 * NHT * KSEQ * KSEQ];  // 512
  __shared__ float act[2 * NHT * KSEQ];        // 64
  __shared__ float hid[2 * 64];
  __shared__ float sig3[2][3];
  const int t = threadIdx.x, lane = t & 31, wave = t >> 5;
  const int e0 = blockIdx.x * 2;

  // A tile: rows 0-7 = h_seq[src[e0]], rows 8-15 = h_seq[src[e0+1]]
  for (int j = t; j < 512; j += 256) {
    int r = j >> 5, c4 = j & 31;
    const float4* sr =
        (const float4*)(hseq + ((size_t)ei[e0 + (r >> 3)] * KSEQ + (r & 7)) * DIM);
    *(h4v*)(at + 4 * j) = f4toh4(sr[c4]);
  }
  __syncthreads();
  {
    v8f q = tile_gemm_p(at, ptsWq, wave, lane);
    const int n = (wave << 4) + (lane & 15), hi = lane >> 4;
#pragma unroll
    for (int v = 0; v < 8; ++v) qt[(v + (hi << 3)) * DIM + n] = q[v] + tsbq[n];
  }
  __syncthreads();
  for (int j = t; j < 512; j += 256) {
    int r = j >> 5, c4 = j & 31;
    const float4* sr = (const float4*)(hseq +
        ((size_t)ei[NEDGES + e0 + (r >> 3)] * KSEQ + (r & 7)) * DIM);
    *(h4v*)(at + 4 * j) = f4toh4(sr[c4]);
  }
  __syncthreads();
  {
    v8f kk = tile_gemm_p(at, ptsWk, wave, lane);
    const int n = (wave << 4) + (lane & 15), hi = lane >> 4;
#pragma unroll
    for (int v = 0; v < 8; ++v) kt[(v + (hi << 3)) * DIM + n] = kk[v] + tsbk[n];
  }
  __syncthreads();

  // scores s[e,h,k,l], scale 32^-0.5
  for (int p = t; p < 512; p += 256) {
    int eidx = p >> 8, rem = p & 255;
    int h = rem >> 6, kk = (rem >> 3) & 7, l = rem & 7;
    const float* qr = &qt[(eidx * 8 + kk) * DIM + h * HDT];
    const float* kr = &kt[(eidx * 8 + l) * DIM + h * HDT];
    float s = 0.f;
#pragma unroll
    for (int d = 0; d < HDT; ++d) s += qr[d] * kr[d];
    sc[p] = s * 0.17677669529663687f;
  }
  __syncthreads();
  if (t < 64) {  // softmax diag per (e,h,k)
    int eidx = t >> 5, h = (t >> 3) & 3, kk = t & 7;
    const float* row = &sc[((eidx * NHT + h) * KSEQ + kk) * KSEQ];
    float m = row[0];
    for (int l = 1; l < 8; ++l) m = fmaxf(m, row[l]);
    float den = 0.f;
    for (int l = 0; l < 8; ++l) den += expf(row[l] - m);
    act[t] = expf(row[kk] - m) / den;
  }
  if (t < 128) {  // ep hidden layer (128 -> 64) + gelu
    int eidx = t >> 6, u = t & 63;
    const float* er = &edge_emb[(size_t)(e0 + eidx) * DIM];
    float s = epb1[u];
    for (int d = 0; d < DIM; ++d) s += er[d] * epW1[d * 64 + u];
    hid[eidx * 64 + u] = geluf(s);
  }
  if (wave < 2) {  // cosine attr, one wave per edge
    int e = e0 + wave;
    const float* xi = &x[(size_t)ei[e] * DIM];
    const float* xj = &x[(size_t)ei[NEDGES + e] * DIM];
    float dd = 0.f, ni = 0.f, nj = 0.f;
    for (int d = lane; d < DIM; d += 32) {
      float a = xi[d], b = xj[d];
      dd += a * b; ni += a * a; nj += b * b;
    }
#pragma unroll
    for (int off = 16; off > 0; off >>= 1) {
      dd += __shfl_xor(dd, off, 32);
      ni += __shfl_xor(ni, off, 32);
      nj += __shfl_xor(nj, off, 32);
    }
    if (lane == 0) {
      float c = dd / (fmaxf(sqrtf(ni), 1e-8f) * fmaxf(sqrtf(nj), 1e-8f));
      sig3[wave][0] = (c + 1.f) * 0.5f;
    }
  }
  __syncthreads();
  if (t < 2) {
    float a = 0.f;
    for (int i = 0; i < 32; ++i) a += act[t * 32 + i];
    sig3[t][1] = a * (1.f / 32.f);
    float s = epb2[0];
    for (int u = 0; u < 64; ++u) s += hid[t * 64 + u] * epW2[u];
    sig3[t][2] = sigm(s);
    float o = opb2[0];
    for (int j = 0; j < 16; ++j) {
      float hh = opb1[j];
      for (int i3 = 0; i3 < 3; ++i3) hh += sig3[t][i3] * opW1[i3 * 16 + j];
      o += geluf(hh) * opW2[j];
    }
    trust[e0 + t] = sigm(o);
  }
}

// ---------------- segment softmax via atomics ----------------
__global__ __launch_bounds__(256) void k_init(unsigned* maxb, float* den, float* outacc) {
  int t = blockIdx.x * 256 + threadIdx.x;
  if (t < NNODES * DIM) outacc[t] = 0.f;
  if (t < NNODES * HEADS) { maxb[t] = 0u; den[t] = 0.f; }
}

__global__ __launch_bounds__(256) void k_logits(const int* ei, const float* Q,
                                                const float* Kj, const float* trust,
                                                float* logits, unsigned* maxb) {
  int t = blockIdx.x * 256 + threadIdx.x;
  if (t >= NEDGES * HEADS) return;
  int e = t >> 3, h = t & 7;
  int d0 = ei[NEDGES + e];
  const float* qr = &Q[(size_t)d0 * DIM + h * HD];
  const float* kr = &Kj[(size_t)e * DIM + h * HD];
  float s = 0.f;
#pragma unroll
  for (int d = 0; d < HD; ++d) s += qr[d] * kr[d];
  s *= 0.25f * trust[e];  // hd^-0.5 = 16^-0.5
  logits[t] = s;
  atomicMax(&maxb[d0 * HEADS + h], f2s(s));
}

__global__ __launch_bounds__(256) void k_expsum(const int* ei, const float* logits,
                                                const unsigned* maxb, float* expv,
                                                float* den) {
  int t = blockIdx.x * 256 + threadIdx.x;
  if (t >= NEDGES * HEADS) return;
  int e = t >> 3, h = t & 7;
  int d0 = ei[NEDGES + e];
  float v = expf(logits[t] - s2f(maxb[d0 * HEADS + h]));
  expv[t] = v;
  atomicAdd(&den[d0 * HEADS + h], v);
}

__global__ __launch_bounds__(256) void k_accum(const int* ei, const float* Vj,
                                               const float* expv, const float* den,
                                               float* outacc) {
  int t = blockIdx.x * 256 + threadIdx.x;
  if (t >= NEDGES * DIM) return;
  int e = t >> 7, d = t & 127, h = d >> 4;
  int d0 = ei[NEDGES + e];
  float alpha = expv[e * HEADS + h] / den[d0 * HEADS + h];
  atomicAdd(&outacc[(size_t)d0 * DIM + d], Vj[t] * alpha);
}

// ---------------- Kernel: Wo GEMM + gate + LayerNorm ----------------
__global__ __launch_bounds__(256) void k_out(
    const float* outacc, const float* x, const float* gbuf, const _Float16* pWo,
    const float* bo, const float* lng, const float* lnb, float* y) {
  __shared__ __align__(16) _Float16 at[16 * DIM];
  __shared__ float yt[16 * DIM];
  __shared__ float mu[16], rs[16];
  const int t = threadIdx.x, lane = t & 31, wave = t >> 5;
  const int row0 = blockIdx.x * 16;
  const float4* os = (const float4*)(outacc + (size_t)row0 * DIM);
  for (int j = t; j < 512; j += 256) *(h4v*)(at + 4 * j) = f4toh4(os[j]);
  __syncthreads();
  v8f o = tile_gemm_p(at, pWo, wave, lane);
  const int n = (wave << 4) + (lane & 15), hi = lane >> 4;
#pragma unroll
  for (int v = 0; v < 8; ++v) {
    int r = v + (hi << 3);
    float gg = gbuf[row0 + r];
    yt[r * DIM + n] = gg * x[(size_t)(row0 + r) * DIM + n] + (1.f - gg) * (o[v] + bo[n]);
  }
  __syncthreads();
  if (t < 16) {
    float s = 0.f;
    for (int d = 0; d < DIM; ++d) s += yt[t * DIM + d];
    float m = s * (1.f / DIM);
    float vv = 0.f;
    for (int d = 0; d < DIM; ++d) { float df = yt[t * DIM + d] - m; vv += df * df; }
    mu[t] = m; rs[t] = rsqrtf(vv * (1.f / DIM) + 1e-5f);
  }
  __syncthreads();
  for (int i = t; i < 16 * DIM; i += 256) {
    int r = i >> 7, c = i & 127;
    y[(size_t)(row0 + r) * DIM + c] = (yt[i] - mu[r]) * rs[r] * lng[c] + lnb[c];
  }
}

extern "C" void kernel_launch(void* const* d_in, const int* in_sizes, int n_in,
                              void* d_out, int out_size, void* d_ws, size_t ws_size,
                              hipStream_t stream) {
  (void)in_sizes; (void)n_in; (void)out_size; (void)ws_size;
  const float* x        = (const float*)d_in[0];
  const int*   ei       = (const int*)d_in[1];
  const float* edge_emb = (const float*)d_in[2];
  const float* time_emb = (const float*)d_in[3];
  const float* hseq     = (const float*)d_in[4];
  const float* Wq  = (const float*)d_in[5],  *bq  = (const float*)d_in[6];
  const float* Wk  = (const float*)d_in[7],  *bk  = (const float*)d_in[8];
  const float* Wv  = (const float*)d_in[9],  *bv  = (const float*)d_in[10];
  const float* Wo  = (const float*)d_in[11], *bo  = (const float*)d_in[12];
  const float* Wek = (const float*)d_in[13], *bek = (const float*)d_in[14];
  const float* Wev = (const float*)d_in[15], *bev = (const float*)d_in[16];
  const float* Wtk = (const float*)d_in[17], *btk = (const float*)d_in[18];
  const float* tsWq = (const float*)d_in[19], *tsbq = (const float*)d_in[20];
  const float* tsWk = (const float*)d_in[21], *tsbk = (const float*)d_in[22];
  const float* epW1 = (const float*)d_in[23], *epb1 = (const float*)d_in[24];
  const float* epW2 = (const float*)d_in[25], *epb2 = (const float*)d_in[26];
  const float* opW1 = (const float*)d_in[27], *opb1 = (const float*)d_in[28];
  const float* opW2 = (const float*)d_in[29], *opb2 = (const float*)d_in[30];
  const float* Wg   = (const float*)d_in[31], *bg   = (const float*)d_in[32];
  const float* lng  = (const float*)d_in[33], *lnb  = (const float*)d_in[34];
  float* y = (float*)d_out;

  const size_t ND = (size_t)NNODES * DIM, ED = (size_t)NEDGES * DIM;
  float* w      = (float*)d_ws;
  float* Q      = w;               // N*DIM
  float* Kb     = Q + ND;          // N*DIM
  float* Vb     = Kb + ND;         // N*DIM
  float* gbuf   = Vb + ND;         // N
  float* Kj     = gbuf + NNODES;   // E*DIM
  float* Vj     = Kj + ED;         // E*DIM
  float* trust  = Vj + ED;         // E
  float* logits = trust + NEDGES;  // E*HEADS
  float* expv   = logits + (size_t)NEDGES * HEADS;
  unsigned* maxb = (unsigned*)(expv + (size_t)NEDGES * HEADS);  // N*HEADS
  float* den    = (float*)(maxb + (size_t)NNODES * HEADS);      // N*HEADS
  float* outacc = den + (size_t)NNODES * HEADS;                 // N*DIM
  _Float16* wpack = (_Float16*)(outacc + ND);  // 9 * 16384 halves (288 KB)

  k_pack<<<(NWPACK * WPELEM) / 256, 256, 0, stream>>>(Wq, Wk, Wv, Wo, Wek, Wev,
                                                      Wtk, tsWq, tsWk, wpack);
  const _Float16* pWq   = wpack + 0 * WPELEM;
  const _Float16* pWk   = wpack + 1 * WPELEM;
  const _Float16* pWv   = wpack + 2 * WPELEM;
  const _Float16* pWo   = wpack + 3 * WPELEM;
  const _Float16* pWek  = wpack + 4 * WPELEM;
  const _Float16* pWev  = wpack + 5 * WPELEM;
  const _Float16* pWtk  = wpack + 6 * WPELEM;
  const _Float16* ptsWq = wpack + 7 * WPELEM;
  const _Float16* ptsWk = wpack + 8 * WPELEM;

  k_node_qkvg<<<NNODES / 16, 256, 0, stream>>>(x, pWq, bq, pWk, bk, pWv, bv, Wg, bg,
                                               Q, Kb, Vb, gbuf);
  k_edge_kv<<<NEDGES / 16, 256, 0, stream>>>(ei, edge_emb, time_emb, pWek, bek,
                                             pWev, bev, pWtk, btk, Kb, Vb, Kj, Vj);
  k_trust<<<NEDGES / 2, 256, 0, stream>>>(ei, x, edge_emb, hseq, ptsWq, tsbq, ptsWk,
                                          tsbk, epW1, epb1, epW2, epb2, opW1, opb1,
                                          opW2, opb2, trust);
  k_init<<<(NNODES * DIM) / 256, 256, 0, stream>>>(maxb, den, outacc);
  k_logits<<<(NEDGES * HEADS) / 256, 256, 0, stream>>>(ei, Q, Kj, trust, logits, maxb);
  k_expsum<<<(NEDGES * HEADS) / 256, 256, 0, stream>>>(ei, logits, maxb, expv, den);
  k_accum<<<(NEDGES * DIM) / 256, 256, 0, stream>>>(ei, Vj, expv, den, outacc);
  k_out<<<NNODES / 16, 256, 0, stream>>>(outacc, x, gbuf, pWo, bo, lng, lnb, y);
}
